// MultiHeadAttention_50568944943685
// MI455X (gfx1250) — compile-verified, matrix operand order checked by reading, not run on hardware
//
#include <hip/hip_runtime.h>
#include <hip/hip_bf16.h>
#include <stdint.h>

// ---------------------------------------------------------------------------
// MI455X (gfx1250) multi-head attention, bf16 WMMA path, f32 accumulate.
// Async global->LDS (ASYNCcnt) double-buffered staging when available.
// ---------------------------------------------------------------------------

typedef __attribute__((ext_vector_type(16))) __bf16 v16bf;
typedef __attribute__((ext_vector_type(8)))  float  v8f;

#define DIM   1024
#define HEADS 16
#define HDIM  64
#define SEQ   2048

#if __has_builtin(__builtin_amdgcn_global_load_async_to_lds_b128) && \
    __has_builtin(__builtin_amdgcn_s_wait_asynccnt)
#define HAVE_ASYNC_LDS 1
#else
#define HAVE_ASYNC_LDS 0
#endif

#if HAVE_ASYNC_LDS
// toolchain signature (from diagnostic): params are pointers to
// GCC-style int4 vectors, AS(1) global source / AS(3) LDS destination.
typedef int v4i_b128 __attribute__((vector_size(4 * sizeof(int))));
typedef __attribute__((address_space(1))) v4i_b128 glb_b128;
typedef __attribute__((address_space(3))) v4i_b128 lds_b128;
// one wave moves 32 lanes x 16B with a single GLOBAL_LOAD_ASYNC_TO_LDS_B128
__device__ __forceinline__ void async_cp16(const __bf16* g, __bf16* l) {
  __builtin_amdgcn_global_load_async_to_lds_b128(
      (glb_b128*)g, (lds_b128*)l, 0, 0);
}
#endif

// ---- WMMA fragment helpers -------------------------------------------------
// 16-bit A-matrix 16x32 layout (ISA 7.12.2): lane L holds row M=L%16;
// vector element e maps to K = 8*(L/16) + (e<8 ? e : 16 + (e-8)).
// The identical pattern serves the B operand when B is stored transposed
// ([N][K] row-major), with L%16 selecting the output column N.
union Frag16 { v16bf v; uint4 q[2]; };

__device__ __forceinline__ v16bf load_frag(const __bf16* base, int stride, int lane) {
  const int r  = lane & 15;
  const int kb = (lane >> 4) << 3;          // 0 or 8
  const __bf16* p = base + r * stride + kb;
  Frag16 f;
  f.q[0] = *(const uint4*)(p);              // K = kb .. kb+7
  f.q[1] = *(const uint4*)(p + 16);         // K = 16+kb .. 16+kb+7
  return f.v;
}

__device__ __forceinline__ v8f wmma_bf16(v16bf a, v16bf b, v8f c) {
  // (neg_a, A, neg_b, B, c_mod, C, reuse_a, reuse_b)
  return __builtin_amdgcn_wmma_f32_16x16x32_bf16(false, a, false, b, (short)0, c,
                                                 false, false);
}

// ---- fp32 -> bf16 elementwise conversion -----------------------------------
__global__ __launch_bounds__(256) void cvt_bf16_kernel(
    const float* __restrict__ in, __bf16* __restrict__ out, unsigned n) {
  unsigned base = (blockIdx.x * blockDim.x + threadIdx.x) * 4u;
  if (base + 3u < n) {
    float4 f = *(const float4*)(in + base);
    out[base + 0] = (__bf16)f.x;
    out[base + 1] = (__bf16)f.y;
    out[base + 2] = (__bf16)f.z;
    out[base + 3] = (__bf16)f.w;
  }
}

// ---- fused transpose + convert for weights: Wt[n][k] = (bf16)W[k][n] -------
__global__ __launch_bounds__(1024) void transpose_cvt_kernel(
    const float* __restrict__ W, __bf16* __restrict__ Wt) {
  __shared__ float t[32][33];
  const int k = blockIdx.y * 32 + threadIdx.y;
  const int n = blockIdx.x * 32 + threadIdx.x;
  t[threadIdx.y][threadIdx.x] = W[(size_t)k * DIM + n];
  __syncthreads();
  const int nn = blockIdx.x * 32 + threadIdx.y;
  const int kk = blockIdx.y * 32 + threadIdx.x;
  Wt[(size_t)nn * DIM + kk] = (__bf16)t[threadIdx.x][threadIdx.y];
}

// ---- GEMM + bias: C[M][N] = A[M][K] * Bt[N][K]^T + bias --------------------
// OUT_MODE: 0 = f32 standard, 1 = bf16 standard,
//           2 = bf16 written transposed per head: Vt[b][h][d][s]
#define BM  128
#define BN  128
#define BK  32
#define LDT 40   // padded K-stride in LDS (32+8): 20-bank row stride -> no conflicts

template<int OUT_MODE>
__global__ __launch_bounds__(256) void gemm_bias_kernel(
    const __bf16* __restrict__ A,    // [M][K]
    const __bf16* __restrict__ Bt,   // [N][K]
    const float*  __restrict__ bias, // [N]
    void* __restrict__ Cout, int M) {
  const int K = DIM, N = DIM;

  const int tid  = threadIdx.x;
  const int lane = tid & 31;
  const int wave = tid >> 5;
  const int bm = blockIdx.x * BM;
  const int bn = blockIdx.y * BN;
  const int wr = (wave >> 1) * 32;   // wave row offset in tile (4 wave-rows)
  const int wc = (wave & 1) * 64;    // wave col offset in tile (2 wave-cols)

  v8f acc[2][4];
  for (int i = 0; i < 2; ++i)
    for (int j = 0; j < 4; ++j)
      for (int e = 0; e < 8; ++e) acc[i][j][e] = 0.0f;

  const int lr = tid >> 1;           // 0..127: tile row this thread stages
  const int lc = (tid & 1) * 16;     // 0/16:  16 bf16 (two b128) per thread

#if HAVE_ASYNC_LDS
  // double-buffered LDS, filled by GLOBAL_LOAD_ASYNC_TO_LDS_B128 (ASYNCcnt)
  __shared__ __bf16 sA[2][BM * LDT];
  __shared__ __bf16 sB[2][BN * LDT];

  const __bf16* gA = A  + (size_t)(bm + lr) * K + lc;
  const __bf16* gB = Bt + (size_t)(bn + lr) * K + lc;

  // prologue: chunk 0 -> buffer 0
  async_cp16(gA,     &sA[0][lr * LDT + lc]);
  async_cp16(gA + 8, &sA[0][lr * LDT + lc + 8]);
  async_cp16(gB,     &sB[0][lr * LDT + lc]);
  async_cp16(gB + 8, &sB[0][lr * LDT + lc + 8]);

  int cur = 0;
  for (int kc = 0; kc < K; kc += BK) {
    __builtin_amdgcn_s_wait_asynccnt(0);  // this wave's fills of buf[cur] done
    __syncthreads();                      // everyone's fills done / readers done
    if (kc + BK < K) {                    // prefetch next chunk during compute
      const __bf16* ga2 = gA + kc + BK;
      const __bf16* gb2 = gB + kc + BK;
      __bf16* la2 = &sA[cur ^ 1][lr * LDT + lc];
      __bf16* lb2 = &sB[cur ^ 1][lr * LDT + lc];
      async_cp16(ga2,     la2);
      async_cp16(ga2 + 8, la2 + 8);
      async_cp16(gb2,     lb2);
      async_cp16(gb2 + 8, lb2 + 8);
    }
    const __bf16* bA = &sA[cur][0];
    const __bf16* bB = &sB[cur][0];
#else
  __shared__ __bf16 sAs[BM * LDT];
  __shared__ __bf16 sBs[BN * LDT];
  for (int kc = 0; kc < K; kc += BK) {
    const uint4* ga = (const uint4*)(A  + (size_t)(bm + lr) * K + kc + lc);
    const uint4* gb = (const uint4*)(Bt + (size_t)(bn + lr) * K + kc + lc);
    uint4 a0 = ga[0], a1 = ga[1];
    uint4 b0 = gb[0], b1 = gb[1];
    __syncthreads();
    *(uint4*)(sAs + lr * LDT + lc)     = a0;
    *(uint4*)(sAs + lr * LDT + lc + 8) = a1;
    *(uint4*)(sBs + lr * LDT + lc)     = b0;
    *(uint4*)(sBs + lr * LDT + lc + 8) = b1;
    __syncthreads();
    const __bf16* bA = sAs;
    const __bf16* bB = sBs;
#endif

    v16bf af0 = load_frag(bA + (wr +  0) * LDT, LDT, lane);
    v16bf af1 = load_frag(bA + (wr + 16) * LDT, LDT, lane);
    v16bf bf0 = load_frag(bB + (wc +  0) * LDT, LDT, lane);
    v16bf bf1 = load_frag(bB + (wc + 16) * LDT, LDT, lane);
    v16bf bf2 = load_frag(bB + (wc + 32) * LDT, LDT, lane);
    v16bf bf3 = load_frag(bB + (wc + 48) * LDT, LDT, lane);

    acc[0][0] = wmma_bf16(af0, bf0, acc[0][0]);
    acc[0][1] = wmma_bf16(af0, bf1, acc[0][1]);
    acc[0][2] = wmma_bf16(af0, bf2, acc[0][2]);
    acc[0][3] = wmma_bf16(af0, bf3, acc[0][3]);
    acc[1][0] = wmma_bf16(af1, bf0, acc[1][0]);
    acc[1][1] = wmma_bf16(af1, bf1, acc[1][1]);
    acc[1][2] = wmma_bf16(af1, bf2, acc[1][2]);
    acc[1][3] = wmma_bf16(af1, bf3, acc[1][3]);
#if HAVE_ASYNC_LDS
    cur ^= 1;
#endif
  }

  // epilogue: C/D layout -> lane%16 = N, VGPR j -> M = j + 8*(lane/16)
  const int cn = lane & 15;
  const int mb = (lane >> 4) * 8;
  for (int i = 0; i < 2; ++i) {
    for (int j = 0; j < 4; ++j) {
      const int gr0 = bm + wr + i * 16 + mb;
      const int gc  = bn + wc + j * 16 + cn;
      const float bb = bias[gc];
      for (int e = 0; e < 8; ++e) {
        const float val = acc[i][j][e] + bb;
        const int gr = gr0 + e;
        if (OUT_MODE == 0) {
          ((float*)Cout)[(size_t)gr * N + gc] = val;
        } else if (OUT_MODE == 1) {
          ((__bf16*)Cout)[(size_t)gr * N + gc] = (__bf16)val;
        } else {
          const int b = gr / SEQ, s = gr % SEQ;
          const int h = gc / HDIM, d = gc % HDIM;
          ((__bf16*)Cout)[((size_t)(b * HEADS + h) * HDIM + d) * SEQ + s] =
              (__bf16)val;
        }
      }
    }
  }
}

// ---- flash attention: 4 waves/block, 16 query rows/wave, 64-key chunks -----
#define KPAD 72  // 36-bank row stride: 16 lanes x b128 cover all 64 banks

__global__ __launch_bounds__(128) void flash_attn_kernel(
    const __bf16* __restrict__ Qb,   // [B][S][DIM]
    const __bf16* __restrict__ Kb,   // [B][S][DIM]
    const __bf16* __restrict__ Vt,   // [B][H][HDIM][S]
    __bf16* __restrict__ Ob) {       // [B][S][DIM]
#if HAVE_ASYNC_LDS
  __shared__ __bf16 sK[2][64 * KPAD];     // [key][d], double buffered
  __shared__ __bf16 sV[2][64 * KPAD];     // [d][key], double buffered
#else
  __shared__ __bf16 sK[1][64 * KPAD];
  __shared__ __bf16 sV[1][64 * KPAD];
#endif
  __shared__ __bf16 sP[4][16 * KPAD];     // per-wave P staging (C->A relayout)

  const int tid = threadIdx.x, lane = tid & 31, wave = tid >> 5;
  const int nqt = SEQ / 64;
  const int qt  = blockIdx.x % nqt;
  const int bh  = blockIdx.x / nqt;
  const int h   = bh % HEADS, b = bh / HEADS;

  const size_t qbase = (size_t)b * SEQ * DIM + h * HDIM;
  const int q0 = qt * 64 + wave * 16;

  // Q fragments: 16 rows x 64 dims -> two 16x32 A-frags, straight from global
  const __bf16* qp = Qb + qbase + (size_t)q0 * DIM;
  v16bf qa0 = load_frag(qp,      DIM, lane);
  v16bf qa1 = load_frag(qp + 32, DIM, lane);

  float mrow[8], lrow[8];
  v8f oacc[4];
  for (int e = 0; e < 8; ++e) { mrow[e] = -3.0e38f; lrow[e] = 0.0f; }
  for (int g = 0; g < 4; ++g)
    for (int e = 0; e < 8; ++e) oacc[g][e] = 0.0f;

  const float sc = 0.125f * 1.44269504088896f;  // 1/sqrt(HDIM) * log2(e)

  const int lr = tid >> 1;            // 0..63: staging row
  const int lc = (tid & 1) * 32;      // 0/32: 32 bf16 (four b128) per thread
  const int cn = lane & 15;
  const int mb = (lane >> 4) * 8;
  __bf16* pw = &sP[wave][0];

  const __bf16* gK = Kb + qbase + (size_t)lr * DIM + lc;
  const __bf16* gV = Vt + ((size_t)(b * HEADS + h) * HDIM + lr) * SEQ + lc;

#if HAVE_ASYNC_LDS
  // prologue: chunk 0 -> buffer 0
  {
    __bf16* lk = &sK[0][lr * KPAD + lc];
    __bf16* lv = &sV[0][lr * KPAD + lc];
    async_cp16(gK,      lk);      async_cp16(gK + 8,  lk + 8);
    async_cp16(gK + 16, lk + 16); async_cp16(gK + 24, lk + 24);
    async_cp16(gV,      lv);      async_cp16(gV + 8,  lv + 8);
    async_cp16(gV + 16, lv + 16); async_cp16(gV + 24, lv + 24);
  }
  int cur = 0;
#endif

  for (int k0 = 0; k0 < SEQ; k0 += 64) {
#if HAVE_ASYNC_LDS
    __builtin_amdgcn_s_wait_asynccnt(0);   // this wave's fills of buf[cur] done
    __syncthreads();                       // all fills visible / readers done
    if (k0 + 64 < SEQ) {                   // prefetch next chunk during compute
      const __bf16* gk2 = gK + (size_t)(k0 + 64) * DIM;
      const __bf16* gv2 = gV + (k0 + 64);
      __bf16* lk2 = &sK[cur ^ 1][lr * KPAD + lc];
      __bf16* lv2 = &sV[cur ^ 1][lr * KPAD + lc];
      async_cp16(gk2,      lk2);      async_cp16(gk2 + 8,  lk2 + 8);
      async_cp16(gk2 + 16, lk2 + 16); async_cp16(gk2 + 24, lk2 + 24);
      async_cp16(gv2,      lv2);      async_cp16(gv2 + 8,  lv2 + 8);
      async_cp16(gv2 + 16, lv2 + 16); async_cp16(gv2 + 24, lv2 + 24);
    }
    const __bf16* bK = &sK[cur][0];
    const __bf16* bV = &sV[cur][0];
#else
    const uint4* gk = (const uint4*)(gK + (size_t)k0 * DIM);
    const uint4* gv = (const uint4*)(gV + k0);
    uint4 ka = gk[0], kb2 = gk[1], kc2 = gk[2], kd = gk[3];
    uint4 va = gv[0], vb  = gv[1], vc  = gv[2], vd = gv[3];
    __syncthreads();
    *(uint4*)(&sK[0][lr * KPAD + lc])      = ka;
    *(uint4*)(&sK[0][lr * KPAD + lc +  8]) = kb2;
    *(uint4*)(&sK[0][lr * KPAD + lc + 16]) = kc2;
    *(uint4*)(&sK[0][lr * KPAD + lc + 24]) = kd;
    *(uint4*)(&sV[0][lr * KPAD + lc])      = va;
    *(uint4*)(&sV[0][lr * KPAD + lc +  8]) = vb;
    *(uint4*)(&sV[0][lr * KPAD + lc + 16]) = vc;
    *(uint4*)(&sV[0][lr * KPAD + lc + 24]) = vd;
    __syncthreads();
    const __bf16* bK = &sK[0][0];
    const __bf16* bV = &sV[0][0];
#endif

    // scores S = Q K^T : 4 key-groups x (2 WMMA over d)
    v8f sf[4];
    for (int ng = 0; ng < 4; ++ng) {
      v8f z;
      for (int e = 0; e < 8; ++e) z[e] = 0.0f;
      v16bf bk0 = load_frag(bK + (ng * 16) * KPAD,      KPAD, lane);
      v16bf bk1 = load_frag(bK + (ng * 16) * KPAD + 32, KPAD, lane);
      z = wmma_bf16(qa0, bk0, z);
      z = wmma_bf16(qa1, bk1, z);
      sf[ng] = z;
    }

    // online softmax, log2 domain
    float cm[8];
    for (int e = 0; e < 8; ++e) {
      float m01 = fmaxf(sf[0][e], sf[1][e]);
      float m23 = fmaxf(sf[2][e], sf[3][e]);
      cm[e] = fmaxf(m01, m23) * sc;
    }
    for (int off = 1; off < 16; off <<= 1)
      for (int e = 0; e < 8; ++e)
        cm[e] = fmaxf(cm[e], __shfl_xor(cm[e], off, 32));

    float alpha[8], rs[8];
    for (int e = 0; e < 8; ++e) {
      const float nm = fmaxf(mrow[e], cm[e]);
      alpha[e] = exp2f(mrow[e] - nm);
      mrow[e]  = nm;
      rs[e]    = 0.0f;
    }

    // P = exp2(S*sc - m); spill to per-wave LDS to re-shape as A-frags
    for (int ng = 0; ng < 4; ++ng)
      for (int e = 0; e < 8; ++e) {
        const float p = exp2f(sf[ng][e] * sc - mrow[e]);
        rs[e] += p;
        pw[(mb + e) * KPAD + ng * 16 + cn] = (__bf16)p;
      }
    for (int off = 1; off < 16; off <<= 1)
      for (int e = 0; e < 8; ++e)
        rs[e] += __shfl_xor(rs[e], off, 32);
    for (int e = 0; e < 8; ++e) lrow[e] = alpha[e] * lrow[e] + rs[e];
    for (int g = 0; g < 4; ++g)
      for (int e = 0; e < 8; ++e) oacc[g][e] *= alpha[e];

    // O += P V : P A-frags (keys 0..31 / 32..63), V^T B-frags per d-group
    v16bf pa0 = load_frag(pw,      KPAD, lane);
    v16bf pa1 = load_frag(pw + 32, KPAD, lane);
    for (int dg = 0; dg < 4; ++dg) {
      v16bf bv0 = load_frag(bV + (dg * 16) * KPAD,      KPAD, lane);
      v16bf bv1 = load_frag(bV + (dg * 16) * KPAD + 32, KPAD, lane);
      oacc[dg] = wmma_bf16(pa0, bv0, oacc[dg]);
      oacc[dg] = wmma_bf16(pa1, bv1, oacc[dg]);
    }
#if HAVE_ASYNC_LDS
    cur ^= 1;
#endif
  }

  // normalize and write O (bf16, standard [B][S][DIM] layout)
  for (int dg = 0; dg < 4; ++dg)
    for (int e = 0; e < 8; ++e) {
      const int s = q0 + mb + e;
      const float val = oacc[dg][e] / lrow[e];
      Ob[qbase + (size_t)s * DIM + dg * 16 + cn] = (__bf16)val;
    }
}

// ---------------------------------------------------------------------------
extern "C" void kernel_launch(void* const* d_in, const int* in_sizes, int n_in,
                              void* d_out, int out_size, void* d_ws, size_t ws_size,
                              hipStream_t stream) {
  const float* values = (const float*)d_in[0];
  const float* keysIn = (const float*)d_in[1];
  const float* query  = (const float*)d_in[2];
  const float* Wq = (const float*)d_in[3];
  const float* bq = (const float*)d_in[4];
  const float* Wk = (const float*)d_in[5];
  const float* bk = (const float*)d_in[6];
  const float* Wv = (const float*)d_in[7];
  const float* bv = (const float*)d_in[8];
  const float* Wo = (const float*)d_in[9];
  const float* bo = (const float*)d_in[10];
  float* out = (float*)d_out;

  const int M = in_sizes[2] / DIM;       // B*S
  const size_t MD = (size_t)M * DIM;
  const int B = M / SEQ;

  char* ws = (char*)d_ws;
  size_t off = 0;
  auto alloc = [&](size_t elts) {
    __bf16* p = (__bf16*)(ws + off);
    off += elts * sizeof(__bf16);
    return p;
  };
  __bf16* xv  = alloc(MD);
  __bf16* xk  = alloc(MD);
  __bf16* xq  = alloc(MD);
  __bf16* wqT = alloc((size_t)DIM * DIM);
  __bf16* wkT = alloc((size_t)DIM * DIM);
  __bf16* wvT = alloc((size_t)DIM * DIM);
  __bf16* woT = alloc((size_t)DIM * DIM);
  __bf16* Qb  = alloc(MD);
  __bf16* Kb  = alloc(MD);
  __bf16* Vt  = alloc(MD);               // [B][H][HDIM][S]
  __bf16* Ob  = alloc(MD);
  (void)ws_size; (void)n_in; (void)out_size;

  // fp32 -> bf16 activations
  const unsigned nconvBlocks = (unsigned)(MD / 4 / 256);
  cvt_bf16_kernel<<<nconvBlocks, 256, 0, stream>>>(values, xv, (unsigned)MD);
  cvt_bf16_kernel<<<nconvBlocks, 256, 0, stream>>>(keysIn, xk, (unsigned)MD);
  cvt_bf16_kernel<<<nconvBlocks, 256, 0, stream>>>(query,  xq, (unsigned)MD);

  // weights: transpose + convert so B operand is K-contiguous
  dim3 tg(DIM / 32, DIM / 32), tb(32, 32);
  transpose_cvt_kernel<<<tg, tb, 0, stream>>>(Wq, wqT);
  transpose_cvt_kernel<<<tg, tb, 0, stream>>>(Wk, wkT);
  transpose_cvt_kernel<<<tg, tb, 0, stream>>>(Wv, wvT);
  transpose_cvt_kernel<<<tg, tb, 0, stream>>>(Wo, woT);

  // projections
  dim3 gg(M / BM, DIM / BN), gb(256);
  gemm_bias_kernel<1><<<gg, gb, 0, stream>>>(xq, wqT, bq, Qb, M);
  gemm_bias_kernel<1><<<gg, gb, 0, stream>>>(xk, wkT, bk, Kb, M);
  gemm_bias_kernel<2><<<gg, gb, 0, stream>>>(xv, wvT, bv, Vt, M);  // head-transposed

  // fused softmax(QK^T/sqrt(d)) V
  flash_attn_kernel<<<B * HEADS * (SEQ / 64), 128, 0, stream>>>(Qb, Kb, Vt, Ob);

  // output projection, f32 straight to d_out
  gemm_bias_kernel<0><<<gg, gb, 0, stream>>>(Ob, woT, bo, out, M);
}
